// MultiHeadAttention_81123342286863
// MI455X (gfx1250) — matrix operand order repaired and checked by reading, not compiled, not run on hardware
//
#include <hip/hip_runtime.h>

typedef __bf16 bf16;
typedef __attribute__((ext_vector_type(16))) __bf16 v16bf;
typedef __attribute__((ext_vector_type(8)))  __bf16 v8bf;
typedef __attribute__((ext_vector_type(4)))  __bf16 v4bf;
typedef __attribute__((ext_vector_type(8)))  float  v8f;

#define N_EMBD   1024
#define N_HEAD   16
#define HEAD_DIM 64
#define BATCH    4
#define SEQ      2048
#define M_TOK    (BATCH * SEQ)   // 8192 tokens

static __device__ __forceinline__ v8f wmma_bf16(v16bf a, v16bf b, v8f c) {
  return __builtin_amdgcn_wmma_f32_16x16x32_bf16(false, a, false, b, (short)0, c,
                                                 false, false);
}

// A fragment 16x32 (MxK) from row-major [rows, ld] bf16 memory.
// lane l: row = l&15, halves 0..7 -> K = 8*(l>>4)..+7, halves 8..15 -> +16.
static __device__ __forceinline__ v16bf load_a16(const bf16* p0, int ld) {
  int lane = threadIdx.x & 31;
  const bf16* p = p0 + (lane & 15) * ld + ((lane >> 4) << 3);
  v8bf lo = *(const v8bf*)(p);
  v8bf hi = *(const v8bf*)(p + 16);
  return __builtin_shufflevector(lo, hi, 0,1,2,3,4,5,6,7,8,9,10,11,12,13,14,15);
}

// B fragment 32x16 (KxN) where memory holds rows of B^T (row n, K contiguous).
// lane l: column n = l&15, halves 0..7 -> K = 8*(l>>4)..+7, halves 8..15 -> +16.
static __device__ __forceinline__ v16bf load_b16(const bf16* p0, int ld) {
  int lane = threadIdx.x & 31;
  const bf16* p = p0 + (lane & 15) * ld + ((lane >> 4) << 3);
  v8bf lo = *(const v8bf*)(p);
  v8bf hi = *(const v8bf*)(p + 16);
  return __builtin_shufflevector(lo, hi, 0,1,2,3,4,5,6,7,8,9,10,11,12,13,14,15);
}

__global__ void cvt_f32_bf16(const float* __restrict__ in, bf16* __restrict__ out,
                             int n4) {
  int i = blockIdx.x * blockDim.x + threadIdx.x;
  if (i >= n4) return;
  float4 f = ((const float4*)in)[i];
  v4bf o = {(bf16)f.x, (bf16)f.y, (bf16)f.z, (bf16)f.w};
  ((v4bf*)out)[i] = o;
}

// QKV projection: [8192,1024] x [3072,1024]^T. Wave tile 64(M) x 64(N).
// Epilogue scatters to Q[B,H,T,D], K[B,H,T,D], V^T[B,H,D,T] (bf16).
__global__ __launch_bounds__(256) void qkv_gemm(const bf16* __restrict__ xb,
                                                const bf16* __restrict__ wb,
                                                bf16* __restrict__ qb,
                                                bf16* __restrict__ kb,
                                                bf16* __restrict__ vtb) {
  const int lane = threadIdx.x & 31;
  const int gw = blockIdx.x * 8 + (threadIdx.x >> 5);
  const int tm = gw & 127;   // 128 m-tiles of 64 rows
  const int tn = gw >> 7;    // 48  n-tiles of 64 cols
  const int m0 = tm << 6;
  const int n0 = tn << 6;
  v8f acc[4][4] = {};
  const bf16* brow = wb + (size_t)n0 * N_EMBD;
#pragma unroll 2
  for (int k = 0; k < N_EMBD; k += 32) {
    v16bf a[4];
#pragma unroll
    for (int i = 0; i < 4; ++i)
      a[i] = load_a16(xb + (size_t)(m0 + (i << 4)) * N_EMBD + k, N_EMBD);
#pragma unroll
    for (int j = 0; j < 4; ++j) {
      v16bf bf = load_b16(brow + (size_t)(j << 4) * N_EMBD + k, N_EMBD);
#pragma unroll
      for (int i = 0; i < 4; ++i) acc[i][j] = wmma_bf16(a[i], bf, acc[i][j]);
    }
  }
  const int sel = n0 >> 10;             // 0=Q 1=K 2=V
  const int h = (n0 & 1023) >> 6;       // head (tile spans one head)
  const int hi = lane >> 4, ln = lane & 15;
#pragma unroll
  for (int im = 0; im < 4; ++im)
#pragma unroll
    for (int j = 0; j < 4; ++j)
#pragma unroll
      for (int v = 0; v < 8; ++v) {
        int m = m0 + (im << 4) + v + (hi << 3);
        int d = (j << 4) + ln;
        int b = m >> 11, t = m & (SEQ - 1);
        bf16 val = (bf16)acc[im][j][v];
        size_t bh = (size_t)(b * N_HEAD + h);
        if (sel == 0)      qb[(bh * SEQ + t) * HEAD_DIM + d] = val;
        else if (sel == 1) kb[(bh * SEQ + t) * HEAD_DIM + d] = val;
        else               vtb[(bh * HEAD_DIM + d) * SEQ + t] = val;
      }
}

// Flash attention: one wave per 32-row q block (two 16-row tiles sharing K
// fragments), streaming 32 keys per step. Row-sums computed by WMMA against a
// ones fragment (ls = alpha*ls + P*1), so only the row-max needs shuffles.
__global__ __launch_bounds__(32) void attn(const bf16* __restrict__ qb,
                                           const bf16* __restrict__ kb,
                                           const bf16* __restrict__ vtb,
                                           bf16* __restrict__ aob) {
  __shared__ __align__(16) bf16 pst[32 * 32];  // two 16x32 P tiles
  const int lane = threadIdx.x;
  const int wt = blockIdx.x & (SEQ / 32 - 1);  // 64 q-blocks per head
  const int bh = blockIdx.x >> 6;
  const int h = bh & (N_HEAD - 1), b = bh >> 4;
  const int q0 = wt << 5;
  const bf16* Q  = qb  + (size_t)bh * SEQ * HEAD_DIM;
  const bf16* K  = kb  + (size_t)bh * SEQ * HEAD_DIM;
  const bf16* Vt = vtb + (size_t)bh * SEQ * HEAD_DIM;
  v16bf qa[2][2];
#pragma unroll
  for (int t = 0; t < 2; ++t) {
    qa[t][0] = load_a16(Q + (size_t)(q0 + (t << 4)) * HEAD_DIM, HEAD_DIM);
    qa[t][1] = load_a16(Q + (size_t)(q0 + (t << 4)) * HEAD_DIM + 32, HEAD_DIM);
  }
  v16bf ones;
#pragma unroll
  for (int i = 0; i < 16; ++i) ones[i] = (bf16)1.0f;
  v8f o[2][4] = {};
  v8f ls[2] = {};
  float mrun[2][8];
#pragma unroll
  for (int t = 0; t < 2; ++t)
#pragma unroll
    for (int v = 0; v < 8; ++v) mrun[t][v] = -1e30f;
  const int hi = lane >> 4, ln = lane & 15;
  const float scale = 0.125f;   // 1/sqrt(64)
  const int nkv = q0 + 32;      // causal bound; always a multiple of 32
  for (int kv = 0; kv < nkv; kv += 32) {
    v16bf k00 = load_b16(K + (size_t)kv * HEAD_DIM, HEAD_DIM);
    v16bf k01 = load_b16(K + (size_t)kv * HEAD_DIM + 32, HEAD_DIM);
    v16bf k10 = load_b16(K + (size_t)(kv + 16) * HEAD_DIM, HEAD_DIM);
    v16bf k11 = load_b16(K + (size_t)(kv + 16) * HEAD_DIM + 32, HEAD_DIM);
    v8f s[2][2];
#pragma unroll
    for (int t = 0; t < 2; ++t) {
      v8f s0 = {}, s1 = {};
      s0 = wmma_bf16(qa[t][0], k00, s0);
      s0 = wmma_bf16(qa[t][1], k01, s0);
      s1 = wmma_bf16(qa[t][0], k10, s1);
      s1 = wmma_bf16(qa[t][1], k11, s1);
      s[t][0] = s0; s[t][1] = s1;
    }
    float p0a[2][8], p1a[2][8], alpha[2][8];
#pragma unroll
    for (int t = 0; t < 2; ++t)
#pragma unroll
      for (int v = 0; v < 8; ++v) {
        int qrow = q0 + (t << 4) + v + (hi << 3);
        float a = s[t][0][v] * scale; if (kv + ln > qrow)      a = -1e30f;
        float c = s[t][1][v] * scale; if (kv + 16 + ln > qrow) c = -1e30f;
        float mx = fmaxf(a, c);
        mx = fmaxf(mx, __shfl_xor(mx, 1, 32));
        mx = fmaxf(mx, __shfl_xor(mx, 2, 32));
        mx = fmaxf(mx, __shfl_xor(mx, 4, 32));
        mx = fmaxf(mx, __shfl_xor(mx, 8, 32));
        float nm = fmaxf(mrun[t][v], mx);
        alpha[t][v] = __expf(mrun[t][v] - nm);
        mrun[t][v] = nm;
        p0a[t][v] = __expf(a - nm);
        p1a[t][v] = __expf(c - nm);
      }
    __syncthreads();  // reshape P: C-layout f32 -> LDS -> A-layout bf16
#pragma unroll
    for (int t = 0; t < 2; ++t)
#pragma unroll
      for (int v = 0; v < 8; ++v) {
        int row = (t << 4) + v + (hi << 3);
        pst[row * 32 + ln]      = (bf16)p0a[t][v];
        pst[row * 32 + 16 + ln] = (bf16)p1a[t][v];
        float al = alpha[t][v];
        o[t][0][v] *= al; o[t][1][v] *= al;
        o[t][2][v] *= al; o[t][3][v] *= al;
        ls[t][v] *= al;
      }
    __syncthreads();
    v16bf vf0 = load_b16(Vt + (size_t)0  * SEQ + kv, SEQ);
    v16bf vf1 = load_b16(Vt + (size_t)16 * SEQ + kv, SEQ);
    v16bf vf2 = load_b16(Vt + (size_t)32 * SEQ + kv, SEQ);
    v16bf vf3 = load_b16(Vt + (size_t)48 * SEQ + kv, SEQ);
#pragma unroll
    for (int t = 0; t < 2; ++t) {
      v16bf pa = load_a16(pst + (t << 4) * 32, 32);
      o[t][0] = wmma_bf16(pa, vf0, o[t][0]);
      o[t][1] = wmma_bf16(pa, vf1, o[t][1]);
      o[t][2] = wmma_bf16(pa, vf2, o[t][2]);
      o[t][3] = wmma_bf16(pa, vf3, o[t][3]);
      ls[t]   = wmma_bf16(pa, ones, ls[t]);   // running row sums
    }
  }
#pragma unroll
  for (int t = 0; t < 2; ++t)
#pragma unroll
    for (int v = 0; v < 8; ++v) {
      float inv = 1.0f / ls[t][v];
      int tok = b * SEQ + q0 + (t << 4) + v + (hi << 3);
      bf16* dst = aob + (size_t)tok * N_EMBD + h * HEAD_DIM + ln;
      dst[0]  = (bf16)(o[t][0][v] * inv);
      dst[16] = (bf16)(o[t][1][v] * inv);
      dst[32] = (bf16)(o[t][2][v] * inv);
      dst[48] = (bf16)(o[t][3][v] * inv);
    }
}

// Output projection + bias: [8192,1024] x [1024,1024]^T -> f32. Wave tile 64x64.
__global__ __launch_bounds__(256) void out_gemm(const bf16* __restrict__ ab,
                                                const bf16* __restrict__ wb,
                                                const float* __restrict__ bias,
                                                float* __restrict__ out) {
  const int lane = threadIdx.x & 31;
  const int gw = blockIdx.x * 8 + (threadIdx.x >> 5);
  const int tm = gw & 127;
  const int tn = gw >> 7;  // 16 n-tiles
  const int m0 = tm << 6, n0 = tn << 6;
  v8f acc[4][4] = {};
#pragma unroll 2
  for (int k = 0; k < N_EMBD; k += 32) {
    v16bf a[4];
#pragma unroll
    for (int i = 0; i < 4; ++i)
      a[i] = load_a16(ab + (size_t)(m0 + (i << 4)) * N_EMBD + k, N_EMBD);
#pragma unroll
    for (int j = 0; j < 4; ++j) {
      v16bf bf = load_b16(wb + (size_t)(n0 + (j << 4)) * N_EMBD + k, N_EMBD);
#pragma unroll
      for (int i = 0; i < 4; ++i) acc[i][j] = wmma_bf16(a[i], bf, acc[i][j]);
    }
  }
  const int hi = lane >> 4, ln = lane & 15;
#pragma unroll
  for (int im = 0; im < 4; ++im)
#pragma unroll
    for (int j = 0; j < 4; ++j) {
      int n = n0 + (j << 4) + ln;
      float bv = bias[n];
#pragma unroll
      for (int v = 0; v < 8; ++v) {
        int m = m0 + (im << 4) + v + (hi << 3);
        out[(size_t)m * N_EMBD + n] = acc[im][j][v] + bv;
      }
    }
}

extern "C" void kernel_launch(void* const* d_in, const int* in_sizes, int n_in,
                              void* d_out, int out_size, void* d_ws, size_t ws_size,
                              hipStream_t stream) {
  const float* x     = (const float*)d_in[0];
  const float* w_qkv = (const float*)d_in[1];
  const float* w_out = (const float*)d_in[2];
  const float* b_out = (const float*)d_in[3];
  float* out = (float*)d_out;

  char* ws = (char*)d_ws;
  size_t off = 0;
  bf16* xb    = (bf16*)(ws + off); off += (size_t)M_TOK * N_EMBD * 2;
  bf16* wqkvb = (bf16*)(ws + off); off += (size_t)3 * N_EMBD * N_EMBD * 2;
  bf16* woutb = (bf16*)(ws + off); off += (size_t)N_EMBD * N_EMBD * 2;
  bf16* qb    = (bf16*)(ws + off); off += (size_t)M_TOK * N_EMBD * 2;
  bf16* kbuf  = (bf16*)(ws + off); off += (size_t)M_TOK * N_EMBD * 2;
  bf16* vtb   = (bf16*)(ws + off); off += (size_t)M_TOK * N_EMBD * 2;
  bf16* aob   = (bf16*)(ws + off); off += (size_t)M_TOK * N_EMBD * 2;

  int n4;
  n4 = M_TOK * N_EMBD / 4;
  cvt_f32_bf16<<<(n4 + 255) / 256, 256, 0, stream>>>(x, xb, n4);
  n4 = 3 * N_EMBD * N_EMBD / 4;
  cvt_f32_bf16<<<(n4 + 255) / 256, 256, 0, stream>>>(w_qkv, wqkvb, n4);
  n4 = N_EMBD * N_EMBD / 4;
  cvt_f32_bf16<<<(n4 + 255) / 256, 256, 0, stream>>>(w_out, woutb, n4);

  qkv_gemm<<<768, 256, 0, stream>>>(xb, wqkvb, qb, kbuf, vtb);
  attn<<<BATCH * N_HEAD * (SEQ / 32), 32, 0, stream>>>(qb, kbuf, vtb, aob);
  out_gemm<<<256, 256, 0, stream>>>(aob, woutb, b_out, out);
}
